// GcnEdgeAngle1dPQA_dueling_57097295233388
// MI455X (gfx1250) — compile-verified, matrix operand order checked by reading, not run on hardware
//
#include <hip/hip_runtime.h>
#include <hip/hip_bf16.h>

// ---------------------------------------------------------------------------
// GcnEdgeAngle1dPQA dueling head, CDNA5 (gfx1250) WMMA implementation.
//
// Pipeline (all on `stream`, graph-capture safe):
//   k_init        : x = [node_features, round_n] (f32 + f16 padded), agg = 0
//   k_fold        : fold head pairs (W1@W2, b1@W2+b2) -> 3 dot-vectors
//   conv<NODE>    : per-edge MLP (nc1) via WMMA, atomic scatter-add to agg
//   k_node_update : x = relu(x+agg); refresh xf16; agg = 0
//   conv<EDGE1>   : per-edge MLP (ec1) -> e1 (f16, stride 80)
//   conv<NODE>    : nc2 -> agg
//   k_node_update
//   conv<EDGE2>   : per-edge MLP (ec2, 3 K-segments) -> feat (f16, stride 96)
//                   + append edge_features_1d
//   k_heads       : folded linear heads + truncated-normal MC terms -> 6 outputs
//
// Fragment I/O is fully vectorized: A frags are 2x ds_load_b128 from the
// row-major activation tile; B frags are 2x ds_load_b128 from weights that
// are staged into LDS pre-swizzled into the ISA 7.12.2 fragment layout.
// ---------------------------------------------------------------------------

typedef _Float16 half_t;
typedef __attribute__((ext_vector_type(16))) _Float16 v16h;
typedef __attribute__((ext_vector_type(8)))  _Float16 v8h;
typedef __attribute__((ext_vector_type(8)))  float    v8f;

static constexpr int kN    = 20000;    // nodes
static constexpr int kEund = 200000;   // undirected edges
static constexpr int kE    = 400000;   // directed edges
static constexpr int kD    = 65;       // node feature dim (64 + round channel)

static constexpr int BR   = 128;       // edge rows per workgroup
static constexpr int KPAD = 96;        // padded K per segment (<=66 -> 96)
static constexpr int NPAD = 80;        // padded output width (65 -> 80)
static constexpr int NT   = NPAD / 16; // 5 col tiles
static constexpr int KT   = KPAD / 32; // 3 k tiles
static constexpr float P_SIGMA = 0.2f;

struct MlpW { const float* W[6]; const float* b[6]; };

__device__ inline v16h cat8(v8h lo, v8h hi) {
  return __builtin_shufflevector(lo, hi, 0, 1, 2, 3, 4, 5, 6, 7,
                                 8, 9, 10, 11, 12, 13, 14, 15);
}

__device__ inline v8h zero8h() {
  v8h z = {(_Float16)0, (_Float16)0, (_Float16)0, (_Float16)0,
           (_Float16)0, (_Float16)0, (_Float16)0, (_Float16)0};
  return z;
}

__device__ inline v8f zero8f() {
  v8f z = {0.f, 0.f, 0.f, 0.f, 0.f, 0.f, 0.f, 0.f};
  return z;
}

// A 16x32 f16 frag (ISA 7.12.2): lane's halves are two contiguous 8-half runs
// at k = hsel*8 and k = 16 + hsel*8 within the 32-wide K tile -> 2x b128.
__device__ inline v16h load_frag_A(const half_t* __restrict__ p, int stride, int lane) {
  const int hsel = (lane >> 4) & 1;
  const half_t* row = p + (size_t)(lane & 15) * stride + (hsel ? 8 : 0);
  v8h lo = *(const v8h*)row;
  v8h hi = *(const v8h*)(row + 16);
  return cat8(lo, hi);
}

// --------------------------- init / node update ----------------------------
__global__ __launch_bounds__(256) void k_init(const float* __restrict__ nf,
                                              const int* __restrict__ round_ptr,
                                              float* __restrict__ x32,
                                              float* __restrict__ agg,
                                              half_t* __restrict__ xf16) {
  int t = blockIdx.x * 256 + threadIdx.x;
  if (t >= kN * NPAD) return;
  int n = t / NPAD, c = t % NPAD;
  float rv = (float)round_ptr[0];
  if (c < kD) {
    float v = (c < 64) ? nf[(size_t)n * 64 + c] : rv;
    x32[(size_t)n * kD + c] = v;
    agg[(size_t)n * kD + c] = 0.f;
    xf16[(size_t)n * NPAD + c] = (half_t)v;
  } else {
    xf16[(size_t)n * NPAD + c] = (half_t)0;
  }
}

__global__ __launch_bounds__(256) void k_node_update(float* __restrict__ x32,
                                                     float* __restrict__ agg,
                                                     half_t* __restrict__ xf16) {
  int t = blockIdx.x * 256 + threadIdx.x;
  if (t >= kN * kD) return;
  int n = t / kD, c = t % kD;
  float v = fmaxf(x32[t] + agg[t], 0.f);
  x32[t] = v;
  agg[t] = 0.f;
  xf16[(size_t)n * NPAD + c] = (half_t)v;
}

// ------------------------------- head folding ------------------------------
// fold[h*80 + c] : c<K -> sum_j W1[c][j]*W2[j] ; slot 74 -> b1@W2 + b2
__global__ __launch_bounds__(256) void k_fold(
    const float* p1W, const float* p1b, const float* p2W, const float* p2b,
    const float* v1W, const float* v1b, const float* v2W, const float* v2b,
    const float* a1W, const float* a1b, const float* a2W, const float* a2b,
    float* __restrict__ fold) {
  int t = blockIdx.x * 256 + threadIdx.x;
  if (t >= 3 * 75) return;
  int h = t / 75, c = t % 75;
  const float *W1, *b1, *W2, *b2;
  int K;
  if (h == 0)      { W1 = p1W; b1 = p1b; W2 = p2W; b2 = p2b; K = 73; }
  else if (h == 1) { W1 = v1W; b1 = v1b; W2 = v2W; b2 = v2b; K = 73; }
  else             { W1 = a1W; b1 = a1b; W2 = a2W; b2 = a2b; K = 74; }
  float s = 0.f;
  if (c < K) {
    for (int j = 0; j < 256; ++j) s = fmaf(W1[(size_t)c * 256 + j], W2[j], s);
    fold[h * 80 + c] = s;
  } else if (c == 74) {
    for (int j = 0; j < 256; ++j) s = fmaf(b1[j], W2[j], s);
    fold[h * 80 + 74] = s + b2[0];
  } else {
    fold[h * 80 + c] = 0.f;   // c==73 unused for p/v heads
  }
}

// --------------------------- fused conv MLP kernel -------------------------
// MODE 0: NodeConv  (1 seg : [x[src], angle], K=66)  -> atomicAdd into agg
// MODE 1: EdgeConv1 (2 segs: x[src], x[dst], K=130)  -> e1 f16 (stride 80)
// MODE 2: EdgeConv2 (3 segs: x[src], x[dst], e1, 195)-> feat f16 (stride 96)
template <int MODE>
__global__ __launch_bounds__(256) void conv_kernel(
    const half_t* __restrict__ xf16, const half_t* __restrict__ e1in,
    const int* __restrict__ eidx, const float* __restrict__ angles,
    const float* __restrict__ ew, const float* __restrict__ ef1d,
    float* __restrict__ agg, half_t* __restrict__ e1out,
    half_t* __restrict__ feat, MlpW mp) {
  __shared__ __align__(16) half_t act[2][BR][KPAD];       // 2 x 24 KB
  __shared__ __align__(16) half_t Wf[KT][NT][32][16];     // 15 KB, frag layout
  __shared__ float  bias_s[NPAD];

  const int t = threadIdx.x, wave = t >> 5, lane = t & 31;
  const int ebase = blockIdx.x * BR;

  // zero the K-pad columns of the ping-pong buffer (gather zero-fills act[0])
  for (int i = t; i < BR * 2; i += 256) {
    int r = i >> 1;
    ((v8h*)&act[1][r][NPAD])[i & 1] = zero8h();
  }

  const int NSEG = (MODE == 0) ? 1 : (MODE == 1 ? 2 : 3);
  v8f acc[NT];
  int cur = 0;

  for (int layer = 0; layer < 6; ++layer) {
#pragma unroll
    for (int j = 0; j < NT; ++j) acc[j] = zero8f();
    const int nseg = (layer == 0) ? NSEG : 1;

    for (int seg = 0; seg < nseg; ++seg) {
      if (layer == 0) {
        // gather one 80-wide f16 row per edge into act[0] (2 threads / row,
        // 16-byte vector copies)
        int r = t >> 1, ch = t & 1;
        int e = ebase + r;
        if (e >= kE) e = kE - 1;
        const half_t* srow;
        if (seg == 0)              srow = xf16 + (size_t)eidx[e] * NPAD;       // x[src]
        else if (seg == 1)         srow = xf16 + (size_t)eidx[kE + e] * NPAD;  // x[dst]
        else                       srow = e1in + (size_t)e * NPAD;             // e1
        __builtin_prefetch(srow, 0, 1);                  // global_prefetch_b8
        v8h* drow = (v8h*)&act[0][r][0];
        const v8h* s8 = (const v8h*)srow;
        if (ch == 0) {
#pragma unroll
          for (int j = 0; j < 6; ++j) drow[j] = s8[j];   // cols 0..47
        } else {
#pragma unroll
          for (int j = 6; j < 10; ++j) drow[j] = s8[j];  // cols 48..79
          drow[10] = zero8h();                           // cols 80..95
          drow[11] = zero8h();
          if (MODE == 0) act[0][r][kD] = (half_t)angles[e];
        }
      }

      int krows, koff;
      if (layer == 0) {
        if (MODE == 0) { krows = 66; koff = 0; }
        else           { krows = 65; koff = seg * 65; }
      } else { krows = 65; koff = 0; }

      // Stage W into LDS pre-swizzled into B-fragment layout:
      // frag elem i16 of lane (hsel*16+n16) in tile (kt,ct) holds
      // W[kt*32 + i16 + (i16&8?8:0) + hsel*8][ct*16 + n16]. We iterate the
      // (k,n) space row-major (coalesced global reads) and invert the map.
      const float* Wsrc = mp.W[layer] + (size_t)koff * 65;
      for (int i = t; i < KPAD * NPAD; i += 256) {
        int k = i / NPAD, n = i % NPAD;
        half_t v = (half_t)0;
        if (k < krows && n < 65) v = (half_t)Wsrc[(size_t)k * 65 + n];
        int kt = k >> 5, kk = k & 31;
        int hsel = (kk >> 3) & 1;
        int i16 = (kk & 7) + ((kk >> 4) << 3);
        Wf[kt][n >> 4][(hsel << 4) | (n & 15)][i16] = v;
      }
      if (seg == 0) {
        for (int n = t; n < NPAD; n += 256) bias_s[n] = (n < 65) ? mp.b[layer][n] : 0.f;
      }
      __syncthreads();

      // one 16-row tile per wave, 5 col tiles, K in steps of 32
#pragma unroll
      for (int kt = 0; kt < KT; ++kt) {
        v16h af = load_frag_A(&act[cur][wave * 16][kt * 32], KPAD, lane);
#pragma unroll
        for (int ct = 0; ct < NT; ++ct) {
          const v8h* wp = (const v8h*)&Wf[kt][ct][lane][0];
          v16h bf = cat8(wp[0], wp[1]);
          acc[ct] = __builtin_amdgcn_wmma_f32_16x16x32_f16(
              false, af, false, bf, (short)0, acc[ct], false, false);
        }
      }
      __syncthreads();
    }

    // epilogue: C/D layout -> row = wave*16 + 8*(lane>>4) + j, col = ct*16 + (lane&15)
    const int hsel = lane >> 4, nlane = lane & 15;
    const int nxt = cur ^ 1;
    if (layer < 5) {
#pragma unroll
      for (int ct = 0; ct < NT; ++ct) {
        int col = ct * 16 + nlane;
#pragma unroll
        for (int i = 0; i < 8; ++i) {
          int row = wave * 16 + 8 * hsel + i;
          float v = fmaxf(acc[ct][i] + bias_s[col], 0.f);
          act[nxt][row][col] = (half_t)v;
        }
      }
      cur = nxt;
    } else {
#pragma unroll
      for (int ct = 0; ct < NT; ++ct) {
        int col = ct * 16 + nlane;
#pragma unroll
        for (int i = 0; i < 8; ++i) {
          int row = wave * 16 + 8 * hsel + i;
          int e = ebase + row;
          if (e >= kE) continue;
          float v = acc[ct][i] + bias_s[col];
          if (MODE == 0) {
            if (col < 65) {
              int dn = eidx[kE + e];
              atomicAdd(&agg[(size_t)dn * kD + col], v);
            }
          } else if (MODE == 1) {
            float w = ew[e - ((e >= kEund) ? kEund : 0)];
            e1out[(size_t)e * NPAD + col] = (half_t)((col < 65) ? v * w : 0.f);
          } else {
            if (col < 65) {
              float w = ew[e - ((e >= kEund) ? kEund : 0)];
              feat[(size_t)e * 96 + col] = (half_t)(v * w);
            }
          }
        }
      }
    }
    __syncthreads();
  }

  if (MODE == 2) {
    // append edge_features_1d (cols 65..72) and zero cols 73..95
    for (int i = t; i < BR * 31; i += 256) {
      int r = i / 31, c = 65 + i % 31;
      int e = ebase + r;
      if (e >= kE) continue;
      feat[(size_t)e * 96 + c] =
          (c < 73) ? (half_t)ef1d[(size_t)e * 8 + (c - 65)] : (half_t)0;
    }
  }
}

// ------------------------- truncated-normal helpers ------------------------
__device__ inline float phi_cdf(float x) {
  return 0.5f * erff(x * 0.70710678118654752f) + 0.5f;
}

__device__ inline float ndtri_f(float p) {  // Acklam rational approximation
  p = fminf(fmaxf(p, 1e-7f), 1.f - 1e-7f);
  const float plow = 0.02425f, phigh = 1.f - 0.02425f;
  float q, r, x;
  if (p < plow) {
    q = sqrtf(-2.f * logf(p));
    x = (((((-7.784894002430293e-03f * q - 3.223964580411365e-01f) * q -
            2.400758277161838e+00f) * q - 2.549732539343734e+00f) * q +
          4.374664141464968e+00f) * q + 2.938163982698783e+00f) /
        ((((7.784695709041462e-03f * q + 3.224671290700398e-01f) * q +
           2.445134137142996e+00f) * q + 3.754408661907416e+00f) * q + 1.f);
  } else if (p <= phigh) {
    q = p - 0.5f; r = q * q;
    x = (((((-3.969683028665376e+01f * r + 2.209460984245205e+02f) * r -
            2.759285104469687e+02f) * r + 1.383577518672690e+02f) * r -
          3.066479806614716e+01f) * r + 2.506628277459239e+00f) * q /
        (((((-5.447609879822406e+01f * r + 1.615858368580409e+02f) * r -
            1.556989798598866e+02f) * r + 6.680131188771972e+01f) * r -
          1.328068155288572e+01f) * r + 1.f);
  } else {
    q = sqrtf(-2.f * logf(1.f - p));
    x = -(((((-7.784894002430293e-03f * q - 3.223964580411365e-01f) * q -
             2.400758277161838e+00f) * q - 2.549732539343734e+00f) * q +
           4.374664141464968e+00f) * q + 2.938163982698783e+00f) /
        ((((7.784695709041462e-03f * q + 3.224671290700398e-01f) * q +
           2.445134137142996e+00f) * q + 3.754408661907416e+00f) * q + 1.f);
  }
  return x;
}

__device__ inline float tsample(float p, float Fa, float dF, float uu) {
  float x = p + P_SIGMA * ndtri_f(Fa + uu * dF);
  return fminf(1.f, fmaxf(0.f, x));
}

// ------------------------------- head kernel -------------------------------
__global__ __launch_bounds__(256) void k_heads(const half_t* __restrict__ feat,
                                               const float* __restrict__ fold,
                                               const float* __restrict__ action,
                                               const float* __restrict__ u,
                                               float* __restrict__ out) {
  int e = blockIdx.x * 256 + threadIdx.x;
  if (e >= kE) return;
  const half_t* f = feat + (size_t)e * 96;
  float sp = fold[0 * 80 + 74], sv = fold[1 * 80 + 74], sa = fold[2 * 80 + 74];
  for (int c = 0; c < 73; ++c) {
    float fv = (float)f[c];
    sp = fmaf(fv, fold[c], sp);
    sv = fmaf(fv, fold[80 + c], sv);
    sa = fmaf(fv, fold[160 + c], sa);
  }
  const float coef = fold[160 + 73];
  const float p = 1.f / (1.f + expf(-sp));
  const float v = sv;
  const float a = sa + coef * action[e];

  const float alpha = (0.f - p) * (1.f / P_SIGMA);
  const float beta  = (1.f - p) * (1.f / P_SIGMA);
  const float Fa = phi_cdf(alpha);
  const float dF = phi_cdf(beta) - Fa;

  float msum = 0.f;
#pragma unroll
  for (int s = 0; s < 4; ++s) msum += tsample(p, Fa, dF, u[(size_t)s * kE + e]);
  const float exp_adv = sa + coef * (msum * 0.25f);
  const float q = v + a - exp_adv;
  const float sl = tsample(p, Fa, dF, u[(size_t)4 * kE + e]);
  const float qp = v + (sa + coef * sl) - exp_adv;

  out[e] = p;
  out[(size_t)kE + e] = q;
  out[(size_t)2 * kE + e] = v;
  out[(size_t)3 * kE + e] = a;
  out[(size_t)4 * kE + e] = sl;
  out[(size_t)5 * kE + e] = qp;
}

// ------------------------------- launcher ----------------------------------
extern "C" void kernel_launch(void* const* d_in, const int* in_sizes, int n_in,
                              void* d_out, int out_size, void* d_ws, size_t ws_size,
                              hipStream_t stream) {
  (void)in_sizes; (void)out_size; (void)ws_size;
  const float* node_features = (const float*)d_in[0];
  const float* action_behav  = (const float*)d_in[1];
  const float* edge_weights  = (const float*)d_in[2];
  const float* angles        = (const float*)d_in[3];
  const float* ef1d          = (const float*)d_in[4];
  const float* u             = (const float*)d_in[5];
  // params pytree leaves, sorted dict keys: a1,a2,ec1,ec2,nc1,nc2,p1,p2,v1,v2
  const float* a1W = (const float*)d_in[6];
  const float* a1b = (const float*)d_in[7];
  const float* a2W = (const float*)d_in[8];
  const float* a2b = (const float*)d_in[9];
  MlpW ec1, ec2, nc1, nc2;
  for (int l = 0; l < 6; ++l) {
    ec1.W[l] = (const float*)d_in[10 + 2 * l]; ec1.b[l] = (const float*)d_in[11 + 2 * l];
    ec2.W[l] = (const float*)d_in[22 + 2 * l]; ec2.b[l] = (const float*)d_in[23 + 2 * l];
    nc1.W[l] = (const float*)d_in[34 + 2 * l]; nc1.b[l] = (const float*)d_in[35 + 2 * l];
    nc2.W[l] = (const float*)d_in[46 + 2 * l]; nc2.b[l] = (const float*)d_in[47 + 2 * l];
  }
  const float* p1W = (const float*)d_in[58];
  const float* p1b = (const float*)d_in[59];
  const float* p2W = (const float*)d_in[60];
  const float* p2b = (const float*)d_in[61];
  const float* v1W = (const float*)d_in[62];
  const float* v1b = (const float*)d_in[63];
  const float* v2W = (const float*)d_in[64];
  const float* v2b = (const float*)d_in[65];
  const int* edge_index = (const int*)d_in[n_in - 2];
  const int* round_ptr  = (const int*)d_in[n_in - 1];
  float* out = (float*)d_out;

  // workspace carve (256B aligned)
  uintptr_t pw = (uintptr_t)d_ws;
  auto carve = [&](size_t bytes) -> void* {
    void* r = (void*)pw;
    pw += (bytes + 255) & ~(size_t)255;
    return r;
  };
  float*  x32  = (float*)carve((size_t)kN * kD * 4);
  float*  agg  = (float*)carve((size_t)kN * kD * 4);
  half_t* xf16 = (half_t*)carve((size_t)kN * NPAD * 2);
  half_t* e1   = (half_t*)carve((size_t)kE * NPAD * 2);
  half_t* feat = (half_t*)carve((size_t)kE * 96 * 2);
  float*  fold = (float*)carve(3 * 80 * 4);

  const int EB = (kE + BR - 1) / BR;  // 3125 edge blocks

  k_init<<<(kN * NPAD + 255) / 256, 256, 0, stream>>>(node_features, round_ptr,
                                                      x32, agg, xf16);
  k_fold<<<1, 256, 0, stream>>>(p1W, p1b, p2W, p2b, v1W, v1b, v2W, v2b,
                                a1W, a1b, a2W, a2b, fold);

  // nc1
  conv_kernel<0><<<EB, 256, 0, stream>>>(xf16, e1, edge_index, angles,
                                         edge_weights, ef1d, agg, e1, feat, nc1);
  k_node_update<<<(kN * kD + 255) / 256, 256, 0, stream>>>(x32, agg, xf16);
  // ec1
  conv_kernel<1><<<EB, 256, 0, stream>>>(xf16, e1, edge_index, angles,
                                         edge_weights, ef1d, agg, e1, feat, ec1);
  // nc2
  conv_kernel<0><<<EB, 256, 0, stream>>>(xf16, e1, edge_index, angles,
                                         edge_weights, ef1d, agg, e1, feat, nc2);
  k_node_update<<<(kN * kD + 255) / 256, 256, 0, stream>>>(x32, agg, xf16);
  // ec2 (+ef1d append)
  conv_kernel<2><<<EB, 256, 0, stream>>>(xf16, e1, edge_index, angles,
                                         edge_weights, ef1d, agg, e1, feat, ec2);

  k_heads<<<(kE + 255) / 256, 256, 0, stream>>>(feat, fold, action_behav, u, out);
}
